// StaticSparseAttention_34591666602184
// MI455X (gfx1250) — matrix operand order repaired and checked by reading, not compiled, not run on hardware
//
#include <hip/hip_runtime.h>
#include <cstdint>

// ---------------------------------------------------------------------------
// StaticSparseAttention for MI455X (gfx1250, wave32, WMMA + async-to-LDS)
//
// Pipeline (all on `stream`):
//   1) cvt_bf16:  x, w_qkv, w_out  f32 -> bf16            (GEMM operands)
//   2) gemm_bf16_wmma: qkv = x @ w_qkv^T + b_qkv          (v_wmma + async LDS)
//   3) rope2d:    in-place RoPE on q/k patch tokens (f32)
//   4) cls_attn:  CLS token attends over all S keys
//   5) win_attn:  dilated-window attention, wave per (b,h,l)
//   6) gemm_bf16_wmma: out = attn @ w_out^T + b_out
// ---------------------------------------------------------------------------

#define USE_ASYNC_LDS 1   // global_load_async_to_lds_b128 + s_wait_asynccnt

typedef __bf16 bf16_t;
typedef __attribute__((ext_vector_type(16))) __bf16 v16bf;
typedef __attribute__((ext_vector_type(8)))  float  v8f;

// Problem constants (fixed by the reference harness)
#define CB   2
#define CL   8192
#define CS   8193            // NUM_CLS + L
#define CE   512
#define CH   8
#define CD   64
#define CQKVW 1536           // 3*E
#define CM   (CB * CS)       // 16386 rows
#define CSCALE 0.125f        // 1/sqrt(64)

__device__ __forceinline__ bf16_t f2bf(float f) {
    union { float f; unsigned u; } a; a.f = f;
    unsigned r = a.u + 0x7FFFu + ((a.u >> 16) & 1u);   // round-to-nearest-even
    union { unsigned short s; bf16_t b; } c;
    c.s = (unsigned short)(r >> 16);
    return c.b;
}

// 16-byte global -> LDS copy (async on CDNA5; tracked with ASYNCcnt)
__device__ __forceinline__ void copy16_g2l(const bf16_t* g, bf16_t* l) {
#if USE_ASYNC_LDS
    unsigned lo           = (unsigned)(uintptr_t)l;    // flat addr low 32 = LDS offset
    unsigned long long ga = (unsigned long long)(uintptr_t)g;
    asm volatile("global_load_async_to_lds_b128 %0, %1, off"
                 :: "v"(lo), "v"(ga) : "memory");
#else
    *(uint4*)l = *(const uint4*)g;
#endif
}

// ---------------------------------------------------------------- cvt f32->bf16
__global__ __launch_bounds__(256)
void cvt_bf16_kernel(const float* __restrict__ in, bf16_t* __restrict__ out, int n) {
    for (int i = blockIdx.x * 256 + threadIdx.x; i < n; i += gridDim.x * 256)
        out[i] = f2bf(in[i]);
}

// ---------------------------------------------------------------- bf16 WMMA GEMM
//   C[M,N] = A[M,K] @ W[N,K]^T + bias[N]       (A,W bf16 row-major, C f32)
//   Block = 8 waves computes a 128x64 tile; K-steps of 32 double-buffered in
//   LDS via global_load_async_to_lds_b128; each wave runs 4 WMMAs per step.
//   All 10 ds_load_b128 of a step are issued before the WMMAs so the compiler
//   can use graduated s_wait_dscnt instead of full drains.
__global__ __launch_bounds__(256)
void gemm_bf16_wmma(const bf16_t* __restrict__ A, const bf16_t* __restrict__ W,
                    const float* __restrict__ bias, float* __restrict__ C,
                    int Mi, int Ni, int Ki) {
    __shared__ __align__(16) bf16_t As[2][128 * 32];   // 2 x 8KB
    __shared__ __align__(16) bf16_t Bs[2][64 * 32];    // 2 x 4KB

    const int tid  = threadIdx.x;
    const int lane = tid & 31;
    const int wv   = tid >> 5;
    const int n64  = Ni >> 6;
    const int mt   = blockIdx.x / n64;
    const int nt   = blockIdx.x % n64;
    const int m0   = mt * 128;
    const int n0   = nt * 64;

    // --- cooperative copy mapping: A = 512 chunks (2/thread), B = 256 (1/thread)
    const int ar0 = tid >> 2;                 // rows 0..63   (chunk i=0)
    const int ar1 = ar0 + 64;                 // rows 64..127 (chunk i=1)
    const int ac  = (tid & 3) * 8;            // 8 bf16 = 16B sub-chunk
    int agr0 = m0 + ar0; if (agr0 >= Mi) agr0 = Mi - 1;   // clamp tail rows
    int agr1 = m0 + ar1; if (agr1 >= Mi) agr1 = Mi - 1;
    const bf16_t* Ag0 = A + (size_t)agr0 * Ki + ac;
    const bf16_t* Ag1 = A + (size_t)agr1 * Ki + ac;
    const bf16_t* Wg  = W + (size_t)(n0 + ar0) * Ki + ac;
    const int al0 = ar0 * 32 + ac;
    const int al1 = ar1 * 32 + ac;
    const int bl  = ar0 * 32 + ac;

    // --- fragment addressing (ISA 7.12.2):
    // A 16x32: lane&15 = row, lane-half selects K base {0,8} (+16 for vgpr4-7)
    // B 32x16: lane&15 = N col, lane-half selects K base {0,8} (+16 for vgpr4-7)
    const int akb  = (lane & 16) ? 8 : 0;
    const int bkb  = (lane & 16) ? 8 : 0;
    const int lrow = wv * 16 + (lane & 15);
    const int bcol = lane & 15;

    v8f zero = {0.f, 0.f, 0.f, 0.f, 0.f, 0.f, 0.f, 0.f};
    v8f acc[4] = {zero, zero, zero, zero};

    auto compute_step = [&](int c) {
        union { v16bf v; uint4 q[2]; } af;
        union { v16bf v; uint4 q[2]; } bfr[4];
        af.q[0] = *(const uint4*)&As[c][lrow * 32 + akb];
        af.q[1] = *(const uint4*)&As[c][lrow * 32 + 16 + akb];
#pragma unroll
        for (int j = 0; j < 4; ++j) {
            const int br = (j * 16 + bcol) * 32;
            bfr[j].q[0] = *(const uint4*)&Bs[c][br + bkb];
            bfr[j].q[1] = *(const uint4*)&Bs[c][br + bkb + 16];
        }
#pragma unroll
        for (int j = 0; j < 4; ++j)
            acc[j] = __builtin_amdgcn_wmma_f32_16x16x32_bf16(
                false, af.v, false, bfr[j].v, (short)0, acc[j], false, false);
    };

    const int nsteps = Ki >> 5;
    // prologue: stage K-step 0 into buffer 0
    copy16_g2l(Ag0, &As[0][al0]);
    copy16_g2l(Ag1, &As[0][al1]);
    copy16_g2l(Wg,  &Bs[0][bl]);

    int cur = 0;
    for (int s = 0; s < nsteps - 1; ++s) {    // steady state: always prefetch
        const int k1 = (s + 1) << 5;
        copy16_g2l(Ag0 + k1, &As[cur ^ 1][al0]);
        copy16_g2l(Ag1 + k1, &As[cur ^ 1][al1]);
        copy16_g2l(Wg  + k1, &Bs[cur ^ 1][bl]);
#if USE_ASYNC_LDS
        asm volatile("s_wait_asynccnt 0x3" ::: "memory");  // cur done, next in flight
#endif
        __syncthreads();                      // publish buf[cur]
        compute_step(cur);
        __syncthreads();                      // reads done before buffer reuse
        cur ^= 1;
    }
    // epilogue step (no prefetch)
#if USE_ASYNC_LDS
    asm volatile("s_wait_asynccnt 0x0" ::: "memory");
#endif
    __syncthreads();
    compute_step(cur);

    // C/D layout: VGPR r -> row rbase + r; lane&15 -> col
    const int rbase = m0 + wv * 16 + ((lane & 16) ? 8 : 0);
#pragma unroll
    for (int j = 0; j < 4; ++j) {
        const int col = n0 + j * 16 + (lane & 15);
        const float bv = bias ? bias[col] : 0.f;
#pragma unroll
        for (int r = 0; r < 8; ++r) {
            const int row = rbase + r;
            if (row < Mi) C[(size_t)row * Ni + col] = acc[j][r] + bv;
        }
    }
}

// ---------------------------------------------------------------- RoPE-2D (in place)
// block = 256 (= H(8) x 32 pairs), grid = B*L tokens
__global__ __launch_bounds__(256)
void rope2d_kernel(float* __restrict__ qkv, const int* __restrict__ coords) {
    const int token = blockIdx.x;
    const int b = token / CL;
    const int l = token - b * CL;
    const int s = 1 + l;
    const int h = threadIdx.x >> 5;
    const int p = threadIdx.x & 31;              // pair index 0..31 -> dims 2p, 2p+1

    const float cx = (float)coords[((size_t)b * CL + l) * 2 + 0] * 1e-5f;
    const float cy = (float)coords[((size_t)b * CL + l) * 2 + 1] * 1e-5f;
    const int fi = (p < 16) ? p : (p - 16);
    const float invf = __expf(-((float)(2 * fi) / 32.f) * 9.210340371976184f); // ln(1e4)
    const float ang  = ((p < 16) ? cx : cy) * invf;
    const float c = __cosf(ang), sn = __sinf(ang);

    float* base = qkv + ((size_t)(b * CS + s)) * CQKVW + h * CD + 2 * p;
    {   // q
        float x0 = base[0], x1 = base[1];
        base[0] = x0 * c - x1 * sn;
        base[1] = x1 * c + x0 * sn;
    }
    {   // k (offset E within the 3E row)
        float* kb = base + CE;
        float x0 = kb[0], x1 = kb[1];
        kb[0] = x0 * c - x1 * sn;
        kb[1] = x1 * c + x0 * sn;
    }
}

// ---------------------------------------------------------------- CLS attention
// one block (256 thr) per (b,h); scores staged in LDS (33KB << 320KB/WGP)
__global__ __launch_bounds__(256)
void cls_attn_kernel(const float* __restrict__ qkv, bf16_t* __restrict__ attn) {
    __shared__ float sc[CS];
    __shared__ float qsh[CD];
    __shared__ float red[256];
    __shared__ float accsh[4 * CD];

    const int b = blockIdx.x >> 3;
    const int h = blockIdx.x & 7;
    const int tid = threadIdx.x;

    const float* qp = qkv + ((size_t)b * CS) * CQKVW + h * CD;   // s=0, q
    if (tid < CD) qsh[tid] = qp[tid];
    __syncthreads();

    float lmax = -3.0e38f;
    for (int k = tid; k < CS; k += 256) {
        const float* kp = qkv + ((size_t)(b * CS + k)) * CQKVW + CE + h * CD;
        float s = 0.f;
#pragma unroll
        for (int d = 0; d < CD; ++d) s += qsh[d] * kp[d];
        s *= CSCALE;
        sc[k] = s;
        lmax = fmaxf(lmax, s);
    }
    red[tid] = lmax; __syncthreads();
    for (int off = 128; off > 0; off >>= 1) {
        if (tid < off) red[tid] = fmaxf(red[tid], red[tid + off]);
        __syncthreads();
    }
    const float smax = red[0]; __syncthreads();

    float lsum = 0.f;
    for (int k = tid; k < CS; k += 256) {
        const float e = __expf(sc[k] - smax);
        sc[k] = e;
        lsum += e;
    }
    red[tid] = lsum; __syncthreads();
    for (int off = 128; off > 0; off >>= 1) {
        if (tid < off) red[tid] += red[tid + off];
        __syncthreads();
    }
    const float inv = 1.f / red[0];

    const int d = tid & 63;
    const int grp = tid >> 6;
    float part = 0.f;
    for (int k = grp; k < CS; k += 4) {
        const float* vp = qkv + ((size_t)(b * CS + k)) * CQKVW + 2 * CE + h * CD;
        part += sc[k] * vp[d];
    }
    accsh[tid] = part;
    __syncthreads();
    if (tid < CD) {
        const float o = (accsh[tid] + accsh[64 + tid] + accsh[128 + tid] + accsh[192 + tid]) * inv;
        attn[((size_t)b * CS) * CE + h * CD + tid] = f2bf(o);
    }
}

// ---------------------------------------------------------------- windowed attention
// one wave32 per (b,h,l); lanes own dims {lane, lane+32}
__global__ __launch_bounds__(256)
void win_attn_kernel(const float* __restrict__ qkv, bf16_t* __restrict__ attn) {
    const int lane = threadIdx.x & 31;
    const int wid  = blockIdx.x * 8 + (threadIdx.x >> 5);
    const int l  = wid % CL;
    const int bh = wid / CL;
    const int h = bh & 7;
    const int b = bh >> 3;
    const int s = 1 + l;

    const float* qp = qkv + ((size_t)(b * CS + s)) * CQKVW + h * CD;
    const float qa = qp[lane];
    const float qb = qp[lane + 32];

    float sc[10];
    int   ks[10];
#pragma unroll
    for (int i = 0; i < 10; ++i) {
        bool valid;
        int kidx;
        if (i == 0) { kidx = 0; valid = true; }                  // CLS key
        else {
            const int lk = l + (i - 1 - 4) * 2;                  // dilation 2, window 4
            valid = (lk >= 0) && (lk < CL);
            kidx = valid ? (1 + lk) : s;                         // clamp load addr
        }
        ks[i] = kidx;
        const float* kp = qkv + ((size_t)(b * CS + kidx)) * CQKVW + CE + h * CD;
        float p = qa * kp[lane] + qb * kp[lane + 32];
#pragma unroll
        for (int m = 16; m > 0; m >>= 1) p += __shfl_xor(p, m, 32);
        sc[i] = valid ? (p * CSCALE) : -3.0e38f;
    }

    float mx = sc[0];
#pragma unroll
    for (int i = 1; i < 10; ++i) mx = fmaxf(mx, sc[i]);
    float den = 0.f, w[10];
#pragma unroll
    for (int i = 0; i < 10; ++i) { w[i] = __expf(sc[i] - mx); den += w[i]; }
    const float dinv = 1.f / den;

    float oa = 0.f, ob = 0.f;
#pragma unroll
    for (int i = 0; i < 10; ++i) {
        const float* vp = qkv + ((size_t)(b * CS + ks[i])) * CQKVW + 2 * CE + h * CD;
        oa += w[i] * vp[lane];
        ob += w[i] * vp[lane + 32];
    }
    bf16_t* op = attn + ((size_t)(b * CS + s)) * CE + h * CD;
    op[lane]      = f2bf(oa * dinv);
    op[lane + 32] = f2bf(ob * dinv);
}

// ---------------------------------------------------------------- host launcher
extern "C" void kernel_launch(void* const* d_in, const int* in_sizes, int n_in,
                              void* d_out, int out_size, void* d_ws, size_t ws_size,
                              hipStream_t stream) {
    const float* x      = (const float*)d_in[0];
    const int*   coords = (const int*)d_in[1];
    const float* w_qkv  = (const float*)d_in[2];
    const float* b_qkv  = (const float*)d_in[3];
    const float* w_out  = (const float*)d_in[4];
    const float* b_out  = (const float*)d_in[5];
    float* out = (float*)d_out;

    // workspace carve-out (~136 MB total), 256B aligned slices
    char* ws = (char*)d_ws;
    size_t off = 0;
    auto carve = [&](size_t bytes) -> void* {
        void* p = ws + off;
        off += (bytes + 255) & ~(size_t)255;
        return p;
    };
    bf16_t* xbf   = (bf16_t*)carve((size_t)CM * CE * 2);
    bf16_t* wqkvb = (bf16_t*)carve((size_t)3 * CE * CE * 2);
    bf16_t* woutb = (bf16_t*)carve((size_t)CE * CE * 2);
    float*  qkv   = (float*) carve((size_t)CM * CQKVW * 4);
    bf16_t* attnb = (bf16_t*)carve((size_t)CM * CE * 2);

    // 1) operand conversions to bf16
    {
        int n = CM * CE;
        cvt_bf16_kernel<<<(n + 255) / 256, 256, 0, stream>>>(x, xbf, n);
        n = 3 * CE * CE;
        cvt_bf16_kernel<<<(n + 255) / 256, 256, 0, stream>>>(w_qkv, wqkvb, n);
        n = CE * CE;
        cvt_bf16_kernel<<<(n + 255) / 256, 256, 0, stream>>>(w_out, woutb, n);
    }

    // 2) QKV projection: [16386,512] x [1536,512]^T  (128x64 tile per block)
    {
        const int mtiles = (CM + 127) / 128, n64 = CQKVW / 64;
        gemm_bf16_wmma<<<mtiles * n64, 256, 0, stream>>>(
            xbf, wqkvb, b_qkv, qkv, CM, CQKVW, CE);
    }

    // 3) RoPE-2D on q,k patch tokens (in place, f32)
    rope2d_kernel<<<CB * CL, 256, 0, stream>>>(qkv, coords);

    // 4) CLS attention rows (s = 0)
    cls_attn_kernel<<<CB * CH, 256, 0, stream>>>(qkv, attnb);

    // 5) windowed attention rows (s >= 1): B*H*L waves, 8 waves/block
    win_attn_kernel<<<(CB * CH * CL) / 8, 256, 0, stream>>>(qkv, attnb);

    // 6) output projection: [16386,512] x [512,512]^T -> d_out
    {
        const int mtiles = (CM + 127) / 128, n64 = CE / 64;
        gemm_bf16_wmma<<<mtiles * n64, 256, 0, stream>>>(
            attnb, woutb, b_out, out, CM, CE, CE);
    }
    (void)in_sizes; (void)n_in; (void)out_size; (void)ws_size;
}
